// Attention_84430467104818
// MI455X (gfx1250) — matrix operand order repaired and checked by reading, not compiled, not run on hardware
//
#include <hip/hip_runtime.h>
#include <hip/hip_bf16.h>

// ---------------------------------------------------------------------------
// Types / WMMA helpers (CDNA5 gfx1250, wave32)
// ---------------------------------------------------------------------------
typedef __bf16 bf16_t;
typedef __attribute__((ext_vector_type(16))) __bf16 v16bf;
typedef __attribute__((ext_vector_type(8)))  float  v8f;
typedef __attribute__((ext_vector_type(4)))  unsigned int v4u;

#define DIMC 1024
#define NSEQ 2048
#define NH   16
#define DH   64
#define BSZ  2
#define ATTN_SCALE 0.125f  // 64^-0.5

union FragCast { v16bf v; v4u q[2]; };

// A fragment (16x32 bf16, M x K), src row-major with leading dim `ld`.
// ISA layout: lanes 0-15 rows M=0..15 / K in {0..7,16..23}; lanes 16-31 K in {8..15,24..31}.
__device__ __forceinline__ v16bf load_frag_a(const bf16_t* base, int ld) {
  const int lane = threadIdx.x & 31;
  const int m    = lane & 15;
  const int koff = (lane >> 4) << 3;         // 0 or 8
  const bf16_t* p = base + (size_t)m * ld + koff;
  FragCast u;
  u.q[0] = *(const v4u*)(p);                 // K = koff .. koff+7
  u.q[1] = *(const v4u*)(p + 16);            // K = 16+koff .. 23+koff
  return u.v;
}

// B fragment (32x16 bf16, K x N) loaded from the TRANSPOSED operand Bt (N x K,
// row-major, leading dim `ld`). lane = column N; lanes 0-15 K in {0..7,16..23}, lanes 16-31 K in {8..15,24..31}.
__device__ __forceinline__ v16bf load_frag_b(const bf16_t* base, int ld) {
  const int lane = threadIdx.x & 31;
  const int n    = lane & 15;
  const int koff = (lane >> 4) << 3;         // 0 or 8
  const bf16_t* p = base + (size_t)n * ld + koff;
  FragCast u;
  u.q[0] = *(const v4u*)(p);                 // K = koff .. koff+7
  u.q[1] = *(const v4u*)(p + 16);            // K = 16+koff .. 23+koff
  return u.v;
}

__device__ __forceinline__ v8f wmma_bf16(v16bf a, v16bf b, v8f c) {
  return __builtin_amdgcn_wmma_f32_16x16x32_bf16(
      /*neg_a=*/false, a, /*neg_b=*/false, b,
      /*c_mod=*/(short)0, c, /*reuse_a=*/false, /*reuse_b=*/false);
}

// ---------------------------------------------------------------------------
// fp32 -> bf16 conversion (vectorized x4)
// ---------------------------------------------------------------------------
__global__ __launch_bounds__(256) void cvt_f32_bf16(const float* __restrict__ in,
                                                    bf16_t* __restrict__ out, int n4) {
  const int i = blockIdx.x * blockDim.x + threadIdx.x;
  if (i < n4) {
    const float4 v = ((const float4*)in)[i];
    union { bf16_t b[4]; uint2 u; } pk;
    pk.b[0] = (bf16_t)v.x; pk.b[1] = (bf16_t)v.y;
    pk.b[2] = (bf16_t)v.z; pk.b[3] = (bf16_t)v.w;
    ((uint2*)out)[i] = pk.u;
  }
}

// ---------------------------------------------------------------------------
// QKV GEMM: out(m,o) = sum_c x(m,c) * w(o,c) + b(o),  m in [0,4096), o in [0,3072)
// Writes Q,K as [b,h,n,d] bf16 and V transposed as [b,h,d,n] bf16.
// Block = 8 waves -> 128x128 tile; wave -> 32x64 (2x4 WMMA tiles).
// Section (Q/K/V) is provably scalar: sec = blockIdx.x >> 3.
// ---------------------------------------------------------------------------
__global__ __launch_bounds__(256) void gemm_qkv(const bf16_t* __restrict__ A,
                                                const bf16_t* __restrict__ Bt,
                                                const float* __restrict__ bias,
                                                bf16_t* __restrict__ qws,
                                                bf16_t* __restrict__ kws,
                                                bf16_t* __restrict__ vws) {
  __shared__ __align__(16) bf16_t tlds[8][256];  // 16x16 bf16 transpose tile per wave
  const int wave = threadIdx.x >> 5;
  const int lane = threadIdx.x & 31;
  const int row0 = blockIdx.y * 128 + (wave >> 1) * 32;
  const int col0 = blockIdx.x * 128 + (wave & 1) * 64;
  const int sec  = (int)(blockIdx.x >> 3);               // 0=Q,1=K,2=V (SGPR-uniform)
  const int h    = ((int)(blockIdx.x & 7)) * 2 + (wave & 1);  // head of this 64-col span

  v8f acc[2][4] = {};
  for (int k0 = 0; k0 < DIMC; k0 += 32) {
    __builtin_prefetch(A + (size_t)row0 * DIMC + k0 + 32, 0, 3);
    __builtin_prefetch(Bt + (size_t)col0 * DIMC + k0 + 32, 0, 3);
    const v16bf a0 = load_frag_a(A + (size_t)row0 * DIMC + k0, DIMC);
    const v16bf a1 = load_frag_a(A + (size_t)(row0 + 16) * DIMC + k0, DIMC);
    v16bf bfrag[4];
#pragma unroll
    for (int t = 0; t < 4; ++t)
      bfrag[t] = load_frag_b(Bt + (size_t)(col0 + t * 16) * DIMC + k0, DIMC);
#pragma unroll
    for (int t = 0; t < 4; ++t) {
      acc[0][t] = wmma_bf16(a0, bfrag[t], acc[0][t]);
      acc[1][t] = wmma_bf16(a1, bfrag[t], acc[1][t]);
    }
  }

  const int nhalf = lane & 15;
  const int roff  = (lane >> 4) << 3;
  bf16_t* tl = tlds[wave];

#pragma unroll
  for (int i = 0; i < 2; ++i) {
    const int m0 = row0 + i * 16;
    const int bb = m0 >> 11;              // batch index (tiles never straddle)
    const int n0 = m0 & (NSEQ - 1);
    const size_t bhx = (size_t)bb * NH + h;
#pragma unroll
    for (int t = 0; t < 4; ++t) {
      const float bv = bias[col0 + t * 16 + nhalf];
      // Lane owns tile column d=nhalf, rows n=roff..roff+7 -> pack along n.
      union { bf16_t b[8]; v4u q; } pk;
#pragma unroll
      for (int r = 0; r < 8; ++r) pk.b[r] = (bf16_t)(acc[i][t][r] + bv);

      if (sec == 2) {
        // V^T [b,h,d,n]: packed direction is already the n axis -> direct b128.
        *(v4u*)(vws + (bhx * DH + t * 16 + nhalf) * NSEQ + n0 + roff) = pk.q;
      } else {
        // Q/K [b,h,n,d]: transpose via wave-private LDS (T[d][n] staging).
        *(v4u*)(tl + nhalf * 16 + roff) = pk.q;        // one ds_store_b128
        asm volatile("s_wait_dscnt 0" ::: "memory");
        union { bf16_t b[8]; v4u q; } g;
#pragma unroll
        for (int j = 0; j < 8; ++j) g.b[j] = tl[(roff + j) * 16 + nhalf];
        bf16_t* dst = (sec == 0) ? qws : kws;
        *(v4u*)(dst + (bhx * NSEQ + n0 + nhalf) * DH + t * 16 + roff) = g.q;
        asm volatile("s_wait_dscnt 0" ::: "memory");   // reads done before reuse
      }
    }
  }
}

// ---------------------------------------------------------------------------
// Flash attention: one wave = 16 query rows, sweeps 2048 keys in chunks of 32.
// S = Q K^T (4 WMMA), online softmax (cross-lane shfl over 16-lane halves),
// P transposed C-layout -> A-layout via wave-private LDS, O += P V (4 WMMA).
// grid.x = B*H*(N/128) = 512, 8 waves/block.
// ---------------------------------------------------------------------------
__global__ __launch_bounds__(256) void attn_fa(const bf16_t* __restrict__ qws,
                                               const bf16_t* __restrict__ kws,
                                               const bf16_t* __restrict__ vws,
                                               bf16_t* __restrict__ aout) {
  __shared__ __align__(16) bf16_t lds_p[8][16 * 64];  // P tile + output staging
  const int wave  = threadIdx.x >> 5;
  const int lane  = threadIdx.x & 31;
  const int nhalf = lane & 15;
  const int roff  = (lane >> 4) << 3;

  const int qblk = blockIdx.x & 15;
  const int bh   = blockIdx.x >> 4;           // b*NH + h
  const int q0   = qblk * 128 + wave * 16;

  const bf16_t* Q  = qws + (size_t)bh * NSEQ * DH;
  const bf16_t* Kb = kws + (size_t)bh * NSEQ * DH;
  const bf16_t* Vt = vws + (size_t)bh * DH * NSEQ;
  bf16_t* pbuf = lds_p[wave];

  // Q fragments are loop-invariant (16 rows x 64 d = 2 A-frags).
  const v16bf aq0 = load_frag_a(Q + (size_t)q0 * DH, DH);
  const v16bf aq1 = load_frag_a(Q + (size_t)q0 * DH + 32, DH);

  v8f o0 = {}, o1 = {}, o2 = {}, o3 = {};
  float rm[8], rl[8];
#pragma unroll
  for (int r = 0; r < 8; ++r) { rm[r] = -3.0e38f; rl[r] = 0.f; }

  for (int kb = 0; kb < NSEQ; kb += 32) {
    v8f s0 = {}, s1 = {};
    s0 = wmma_bf16(aq0, load_frag_b(Kb + (size_t)kb * DH,             DH), s0);
    s0 = wmma_bf16(aq1, load_frag_b(Kb + (size_t)kb * DH + 32,        DH), s0);
    s1 = wmma_bf16(aq0, load_frag_b(Kb + (size_t)(kb + 16) * DH,      DH), s1);
    s1 = wmma_bf16(aq1, load_frag_b(Kb + (size_t)(kb + 16) * DH + 32, DH), s1);

    float mloc[8];
#pragma unroll
    for (int r = 0; r < 8; ++r) {
      s0[r] = s0[r] * ATTN_SCALE;
      s1[r] = s1[r] * ATTN_SCALE;
      mloc[r] = fmaxf(s0[r], s1[r]);
    }
    // row-max across the 16 lanes holding this row's 16 columns
#pragma unroll
    for (int off = 8; off >= 1; off >>= 1)
#pragma unroll
      for (int r = 0; r < 8; ++r)
        mloc[r] = fmaxf(mloc[r], __shfl_xor(mloc[r], off, 32));

    float p0[8], p1[8], ls[8];
#pragma unroll
    for (int r = 0; r < 8; ++r) {
      const float mn = fmaxf(rm[r], mloc[r]);
      const float al = __expf(rm[r] - mn);
      rm[r] = mn;
      p0[r] = __expf(s0[r] - mn);
      p1[r] = __expf(s1[r] - mn);
      ls[r] = p0[r] + p1[r];
      rl[r] = rl[r] * al;
      o0[r] = o0[r] * al;
      o1[r] = o1[r] * al;
      o2[r] = o2[r] * al;
      o3[r] = o3[r] * al;
    }
#pragma unroll
    for (int off = 8; off >= 1; off >>= 1)
#pragma unroll
      for (int r = 0; r < 8; ++r)
        ls[r] += __shfl_xor(ls[r], off, 32);
#pragma unroll
    for (int r = 0; r < 8; ++r) rl[r] += ls[r];

    // C-frag layout -> row-major 16x32 bf16 P tile in wave-private LDS
#pragma unroll
    for (int r = 0; r < 8; ++r) {
      pbuf[(roff + r) * 32 + nhalf]      = (bf16_t)p0[r];
      pbuf[(roff + r) * 32 + 16 + nhalf] = (bf16_t)p1[r];
    }
    asm volatile("s_wait_dscnt 0" ::: "memory");

    const v16bf ap = load_frag_a(pbuf, 32);  // ds_load_b128 x2
    o0 = wmma_bf16(ap, load_frag_b(Vt + (size_t)0  * NSEQ + kb, NSEQ), o0);
    o1 = wmma_bf16(ap, load_frag_b(Vt + (size_t)16 * NSEQ + kb, NSEQ), o1);
    o2 = wmma_bf16(ap, load_frag_b(Vt + (size_t)32 * NSEQ + kb, NSEQ), o2);
    o3 = wmma_bf16(ap, load_frag_b(Vt + (size_t)48 * NSEQ + kb, NSEQ), o3);
    asm volatile("s_wait_dscnt 0" ::: "memory");  // P reads done before next overwrite
  }

  // ---- epilogue: normalize, stage 16x64 tile T[d][q] in LDS, b128 stores ----
  const int b = bh >> 4;
  const int h = bh & 15;
#pragma unroll
  for (int r = 0; r < 8; ++r) rl[r] = 1.0f / rl[r];

  {
    union { bf16_t bb[8]; v4u q; } pk;
#pragma unroll
    for (int r = 0; r < 8; ++r) pk.bb[r] = (bf16_t)(o0[r] * rl[r]);
    *(v4u*)(pbuf + (0 * 16 + nhalf) * 16 + roff) = pk.q;
#pragma unroll
    for (int r = 0; r < 8; ++r) pk.bb[r] = (bf16_t)(o1[r] * rl[r]);
    *(v4u*)(pbuf + (1 * 16 + nhalf) * 16 + roff) = pk.q;
#pragma unroll
    for (int r = 0; r < 8; ++r) pk.bb[r] = (bf16_t)(o2[r] * rl[r]);
    *(v4u*)(pbuf + (2 * 16 + nhalf) * 16 + roff) = pk.q;
#pragma unroll
    for (int r = 0; r < 8; ++r) pk.bb[r] = (bf16_t)(o3[r] * rl[r]);
    *(v4u*)(pbuf + (3 * 16 + nhalf) * 16 + roff) = pk.q;
  }
  asm volatile("s_wait_dscnt 0" ::: "memory");

  // 128 chunks (16 q-rows x 8 d-chunks); 4 per lane; each chunk = b128 store.
#pragma unroll
  for (int c = 0; c < 4; ++c) {
    const int chunk = lane * 4 + c;
    const int qr = chunk >> 3;          // 0..15
    const int dc = chunk & 7;           // 0..7
    union { bf16_t bb[8]; v4u q; } g;
#pragma unroll
    for (int j = 0; j < 8; ++j) g.bb[j] = pbuf[(dc * 8 + j) * 16 + qr];
    *(v4u*)(aout + (size_t)(b * NSEQ + q0 + qr) * DIMC + h * DH + dc * 8) = g.q;
  }
}

// ---------------------------------------------------------------------------
// Proj GEMM: out(m,o) = sum_c a(m,c) * w(o,c) + b(o), f32 output via LDS
// transpose staging -> global_store_b128 pairs.
// ---------------------------------------------------------------------------
__global__ __launch_bounds__(256) void gemm_proj(const bf16_t* __restrict__ A,
                                                 const bf16_t* __restrict__ Bt,
                                                 const float* __restrict__ bias,
                                                 float* __restrict__ out) {
  __shared__ __align__(16) float tldsf[8][256];  // 16x16 f32 tile per wave
  const int wave = threadIdx.x >> 5;
  const int lane = threadIdx.x & 31;
  const int row0 = blockIdx.y * 128 + (wave >> 1) * 32;
  const int col0 = blockIdx.x * 128 + (wave & 1) * 64;

  v8f acc[2][4] = {};
  for (int k0 = 0; k0 < DIMC; k0 += 32) {
    __builtin_prefetch(A + (size_t)row0 * DIMC + k0 + 32, 0, 3);
    __builtin_prefetch(Bt + (size_t)col0 * DIMC + k0 + 32, 0, 3);
    const v16bf a0 = load_frag_a(A + (size_t)row0 * DIMC + k0, DIMC);
    const v16bf a1 = load_frag_a(A + (size_t)(row0 + 16) * DIMC + k0, DIMC);
    v16bf bfrag[4];
#pragma unroll
    for (int t = 0; t < 4; ++t)
      bfrag[t] = load_frag_b(Bt + (size_t)(col0 + t * 16) * DIMC + k0, DIMC);
#pragma unroll
    for (int t = 0; t < 4; ++t) {
      acc[0][t] = wmma_bf16(a0, bfrag[t], acc[0][t]);
      acc[1][t] = wmma_bf16(a1, bfrag[t], acc[1][t]);
    }
  }

  const int nhalf = lane & 15;
  const int roff  = (lane >> 4) << 3;
  float* tf = tldsf[wave];

#pragma unroll
  for (int i = 0; i < 2; ++i) {
    const int m0 = row0 + i * 16;
#pragma unroll
    for (int t = 0; t < 4; ++t) {
      const float bv = bias[col0 + t * 16 + nhalf];
      union { float f[8]; v4u q[2]; } pk;
#pragma unroll
      for (int r = 0; r < 8; ++r) pk.f[r] = acc[i][t][r] + bv;
      // stage T[d][n] packed along n
      *(v4u*)(tf + nhalf * 16 + roff)     = pk.q[0];
      *(v4u*)(tf + nhalf * 16 + roff + 4) = pk.q[1];
      asm volatile("s_wait_dscnt 0" ::: "memory");
      union { float f[8]; v4u q[2]; } g;
#pragma unroll
      for (int j = 0; j < 8; ++j) g.f[j] = tf[(roff + j) * 16 + nhalf];
      float* dst = out + (size_t)(m0 + nhalf) * DIMC + col0 + t * 16 + roff;
      *(v4u*)(dst)     = g.q[0];
      *(v4u*)(dst + 4) = g.q[1];
      asm volatile("s_wait_dscnt 0" ::: "memory");
    }
  }
}

// ---------------------------------------------------------------------------
// Host-side launch
// ---------------------------------------------------------------------------
extern "C" void kernel_launch(void* const* d_in, const int* in_sizes, int n_in,
                              void* d_out, int out_size, void* d_ws, size_t ws_size,
                              hipStream_t stream) {
  const float* x      = (const float*)d_in[0];
  const float* qkv_w  = (const float*)d_in[1];
  const float* qkv_b  = (const float*)d_in[2];
  const float* proj_w = (const float*)d_in[3];
  const float* proj_b = (const float*)d_in[4];
  float* out = (float*)d_out;

  const size_t M   = (size_t)BSZ * NSEQ;               // 4096
  const size_t HXE = (size_t)BSZ * NH * NSEQ * DH;     // per-Q/K/V buffer elems

  bf16_t* xb    = (bf16_t*)d_ws;                       // 4096x1024
  bf16_t* wqkv  = xb    + M * DIMC;                    // 3072x1024
  bf16_t* wproj = wqkv  + (size_t)3 * DIMC * DIMC;     // 1024x1024
  bf16_t* qws   = wproj + (size_t)DIMC * DIMC;
  bf16_t* kws   = qws   + HXE;
  bf16_t* vws   = kws   + HXE;
  bf16_t* aouts = vws   + HXE;                         // attention out, bf16

  const int n_x = (int)(M * DIMC);
  const int n_w = 3 * DIMC * DIMC;
  const int n_p = DIMC * DIMC;
  cvt_f32_bf16<<<(n_x / 4) / 256, 256, 0, stream>>>(x, xb, n_x / 4);
  cvt_f32_bf16<<<(n_w / 4) / 256, 256, 0, stream>>>(qkv_w, wqkv, n_w / 4);
  cvt_f32_bf16<<<(n_p / 4) / 256, 256, 0, stream>>>(proj_w, wproj, n_p / 4);

  gemm_qkv<<<dim3(3 * DIMC / 128, M / 128), 256, 0, stream>>>(
      xb, wqkv, qkv_b, qws, kws, vws);

  attn_fa<<<BSZ * NH * (NSEQ / 128), 256, 0, stream>>>(qws, kws, vws, aouts);

  gemm_proj<<<dim3(DIMC / 128, M / 128), 256, 0, stream>>>(
      aouts, wproj, proj_b, out);
}
